// NAMClassifier_20590073217759
// MI455X (gfx1250) — matrix operand order repaired and checked by reading, not compiled, not run on hardware
//
#include <hip/hip_runtime.h>
#include <hip/hip_bf16.h>
#include <math.h>

typedef __attribute__((ext_vector_type(16))) _Float16 v16h;
typedef __attribute__((ext_vector_type(8)))  float    v8f;
typedef __attribute__((ext_vector_type(4)))  float    v4f;

#define USE_ASYNC_LDS 1

#define B_ROWS 16384
#define NFEAT  256
#define H1D    64
#define H2D    32
#define FC     64                 // features per block (one feature-group)
#define NFG    (NFEAT / FC)       // 4 feature-groups
#define NBB    32                 // batch blocks per feature-group
#define WAVES  8
#define TILES_PER_WAVE 4          // 1024 tiles / (NBB*WAVES)
#define TILE_M 16

// Dynamic LDS layout (bytes):
//   smB   : FC*2*2*32*16 halves (W2 f16, WMMA B-layout)  @ 0       = 262144
//   smW1h : FC*2*2*16 halves (f16, A-element order)      @ 262144  =   8192
//   smB1h : FC*2*2*16 halves                             @ 270336  =   8192
//   smB2  : FC*32 f32                                    @ 278528  =   8192
//   smW3  : FC*32 f32                                    @ 286720  =   8192
//   misc  : 1 f32 (sum of this group's b3)               @ 294912  =      4
#define SMEM_BYTES 295040

__global__ void __launch_bounds__(256)
nam_main(const float* __restrict__ x,
         const float* __restrict__ W1,
         const float* __restrict__ b1,
         const float* __restrict__ W2,
         const float* __restrict__ b2,
         const float* __restrict__ W3,
         const float* __restrict__ b3,
         float* __restrict__ partial)
{
    extern __shared__ char smemRaw[];
    _Float16* smB   = (_Float16*)smemRaw;
    _Float16* smW1h = (_Float16*)(smemRaw + 262144);
    _Float16* smB1h = smW1h + 4096;
    float*    smB2  = (float*)(smemRaw + 278528);
    float*    smW3  = smB2 + 2048;
    float*    smMisc = smW3 + 2048;

    const int tid = threadIdx.x;
    const int f0  = blockIdx.y * FC;

    // ---- Stage b2 / W3 (bit-exact contiguous copies) via CDNA5 async LDS loads ----
#if USE_ASYNC_LDS
    for (int e = tid; e < FC * H2D; e += 256) {
        uint32_t l2 = (uint32_t)(uintptr_t)(smB2 + e);      // LDS byte offset
        const float* g2 = b2 + (size_t)f0 * H2D + e;
        asm volatile("global_load_async_to_lds_b32 %0, %1, off"
                     :: "v"(l2), "v"(g2) : "memory");
        uint32_t l3 = (uint32_t)(uintptr_t)(smW3 + e);
        const float* g3 = W3 + (size_t)f0 * H2D + e;
        asm volatile("global_load_async_to_lds_b32 %0, %1, off"
                     :: "v"(l3), "v"(g3) : "memory");
    }
    asm volatile("s_wait_asynccnt 0" ::: "memory");
#else
    for (int e = tid; e < FC * H2D; e += 256) {
        smB2[e] = b2[(size_t)f0 * H2D + e];
        smW3[e] = W3[(size_t)f0 * H2D + e];
    }
#endif

    // ---- Stage W2 into LDS as f16 B-fragments for v_wmma_f32_16x16x32_f16 ----
    // B (32x16) in the contiguous k order this code uses, matching A: lane L holds N=(L&15); element j holds K=(L>>4)*16 + j.
    for (int e = tid; e < FC * 2 * 2 * 32 * 16; e += 256) {
        int j  = e & 15;
        int L  = (e >> 4) & 31;
        int t  = (e >> 9) & 1;   // N tile (0: n=0..15, 1: n=16..31)
        int s  = (e >> 10) & 1;  // K step (0: K=0..31, 1: K=32..63)
        int ff = e >> 11;
        int K  = s * 32 + ((L >> 4) * 16) + j;
        int N  = t * 16 + (L & 15);
        smB[e] = (_Float16)W2[((size_t)(f0 + ff) * H1D + K) * H2D + N];
    }
    // W1/b1 pre-converted to f16 in A-element order: A (16x32) in the contiguous k order used here: lane L holds
    // M=(L&15); element j holds K = (L>>4)*16 + j  (+32 for k-step 1).
    for (int e = tid; e < FC * 2 * 2 * 16; e += 256) {
        int j  = e & 15;
        int g  = (e >> 4) & 1;   // lane group (L>>4)
        int s  = (e >> 5) & 1;   // K step
        int ff = e >> 6;
        int K  = s * 32 + g * 16 + j;
        smW1h[e] = (_Float16)W1[(f0 + ff) * H1D + K];
        smB1h[e] = (_Float16)b1[(f0 + ff) * H1D + K];
    }
    if (tid == 0) {
        float s = 0.f;
        for (int ff = 0; ff < FC; ++ff) s += b3[f0 + ff];
        smMisc[0] = s;
    }
    __syncthreads();

    const int wid  = tid >> 5;
    const int lane = tid & 31;
    const int lo   = lane & 15;
    const int hg   = lane >> 4;
    const int gw   = blockIdx.x * WAVES + wid;   // [0, 256)
    const float b3sum = smMisc[0];
    const v16h z16 = {};

    for (int it = 0; it < TILES_PER_WAVE; ++it) {
        const int tile  = it * (NBB * WAVES) + gw;   // [0, 1024), each exactly once
        const int bbase = tile * TILE_M;
        const float* xrow = x + (size_t)(bbase + lo) * NFEAT + f0;

        // Prefetch next tile's x rows (global_prefetch_b8).
        if (it + 1 < TILES_PER_WAVE)
            __builtin_prefetch(xrow + (size_t)(NBB * WAVES) * TILE_M * NFEAT, 0, 1);

        float val[8] = {0.f, 0.f, 0.f, 0.f, 0.f, 0.f, 0.f, 0.f};

        for (int ffb = 0; ffb < FC; ffb += 4) {
            float xq[4];
            *(v4f*)xq = *(const v4f*)(xrow + ffb);   // one global_load_b128 / 4 features

#pragma unroll
            for (int q = 0; q < 4; ++q) {
                const int ff = ffb + q;
                const _Float16 xh = (_Float16)xq[q];

                // Layer 1 entirely in packed f16: a = relu(x*W1 + b1) as the A fragment.
                const v16h w1v0 = *(const v16h*)(smW1h + (((ff * 2 + 0) * 2 + hg) << 4));
                const v16h b1v0 = *(const v16h*)(smB1h + (((ff * 2 + 0) * 2 + hg) << 4));
                const v16h w1v1 = *(const v16h*)(smW1h + (((ff * 2 + 1) * 2 + hg) << 4));
                const v16h b1v1 = *(const v16h*)(smB1h + (((ff * 2 + 1) * 2 + hg) << 4));
                v16h a0 = __builtin_elementwise_max(w1v0 * xh + b1v0, z16);
                v16h a1 = __builtin_elementwise_max(w1v1 * xh + b1v1, z16);

                // Layer 2: 16x64 @ 64x32 via 4 x v_wmma_f32_16x16x32_f16, f32 accum.
                const v16h* B00 = (const v16h*)(smB + ((size_t)((ff * 2 + 0) * 2 + 0) * 32 + lane) * 16);
                const v16h* B10 = (const v16h*)(smB + ((size_t)((ff * 2 + 1) * 2 + 0) * 32 + lane) * 16);
                const v16h* B01 = (const v16h*)(smB + ((size_t)((ff * 2 + 0) * 2 + 1) * 32 + lane) * 16);
                const v16h* B11 = (const v16h*)(smB + ((size_t)((ff * 2 + 1) * 2 + 1) * 32 + lane) * 16);

                v8f c0 = {}, c1 = {};
                c0 = __builtin_amdgcn_wmma_f32_16x16x32_f16(false, a0, false, *B00, (short)0, c0, false, false);
                c0 = __builtin_amdgcn_wmma_f32_16x16x32_f16(false, a1, false, *B10, (short)0, c0, false, false);
                c1 = __builtin_amdgcn_wmma_f32_16x16x32_f16(false, a0, false, *B01, (short)0, c1, false, false);
                c1 = __builtin_amdgcn_wmma_f32_16x16x32_f16(false, a1, false, *B11, (short)0, c1, false, false);

                // Layer 3: relu(h2 + b2) . W3, accumulated per-lane across features.
                // C layout: VGPR r holds row M = r + 8*(lane>>4), col N = lane&15 (+16 for c1).
                const float b2a = smB2[ff * 32 + lo];
                const float b2b = smB2[ff * 32 + 16 + lo];
                const float w3a = smW3[ff * 32 + lo];
                const float w3b = smW3[ff * 32 + 16 + lo];
#pragma unroll
                for (int r = 0; r < 8; ++r) {
                    val[r] += fmaxf(c0[r] + b2a, 0.f) * w3a
                            + fmaxf(c1[r] + b2b, 0.f) * w3b;
                }
            }
        }

        // One cross-lane reduction per tile (sum over the 16 N-lanes per half-wave).
#pragma unroll
        for (int off = 8; off >= 1; off >>= 1) {
#pragma unroll
            for (int r = 0; r < 8; ++r)
                val[r] += __shfl_xor(val[r], off, 32);
        }
        if (lo == 0) {
#pragma unroll
            for (int r = 0; r < 8; ++r)
                partial[(size_t)blockIdx.y * B_ROWS + bbase + hg * 8 + r] = val[r] + b3sum;
        }
    }
}

__global__ void __launch_bounds__(256)
nam_finalize(const float* __restrict__ partial,
             const float* __restrict__ bias,
             float* __restrict__ out)
{
    int b = blockIdx.x * blockDim.x + threadIdx.x;
    if (b < B_ROWS) {
        float s = bias[0];
#pragma unroll
        for (int fg = 0; fg < NFG; ++fg)
            s += partial[(size_t)fg * B_ROWS + b];
        out[b] = s;                                   // logits
        out[B_ROWS + b] = 1.0f / (1.0f + __expf(-s)); // probs
    }
}

extern "C" void kernel_launch(void* const* d_in, const int* in_sizes, int n_in,
                              void* d_out, int out_size, void* d_ws, size_t ws_size,
                              hipStream_t stream) {
    (void)in_sizes; (void)n_in; (void)out_size; (void)ws_size;
    const float* x    = (const float*)d_in[0];
    const float* W1   = (const float*)d_in[1];
    const float* b1   = (const float*)d_in[2];
    const float* W2   = (const float*)d_in[3];
    const float* b2   = (const float*)d_in[4];
    const float* W3   = (const float*)d_in[5];
    const float* b3   = (const float*)d_in[6];
    const float* bias = (const float*)d_in[7];
    float* out     = (float*)d_out;
    float* partial = (float*)d_ws;   // NFG * B_ROWS floats = 256 KB

    hipFuncSetAttribute(reinterpret_cast<const void*>(nam_main),
                        hipFuncAttributeMaxDynamicSharedMemorySize, SMEM_BYTES);

    dim3 grid(NBB, NFG, 1);
    nam_main<<<grid, 256, SMEM_BYTES, stream>>>(x, W1, b1, W2, b2, W3, b3, partial);
    nam_finalize<<<(B_ROWS + 255) / 256, 256, 0, stream>>>(partial, bias, out);
}
